// RoIHeads_11562051961008
// MI455X (gfx1250) — compile-verified, matrix operand order checked by reading, not run on hardware
//
#include <hip/hip_runtime.h>
#include <hip/hip_bf16.h>

// ---------------- problem constants (match reference) ----------------
constexpr int B = 16;          // images
constexpr int P = 16000;       // proposals per image
constexpr int G = 256;         // gt boxes per image
constexpr int M = P + G;       // proposals + gt (16256)
constexpr int S = 512;         // sampled per image
constexpr int NPOS_MAX = 128;  // S * 0.25
constexpr int C = 91;          // classes
constexpr float BG_T = 0.5f;
constexpr float BETA = 1.0f / 9.0f;

typedef __attribute__((ext_vector_type(16))) _Float16 v16h;
typedef __attribute__((ext_vector_type(8)))  float    v8f;
typedef __attribute__((ext_vector_type(4)))  unsigned int u32x4;
typedef __attribute__((ext_vector_type(8)))  int          i32x8;
typedef __attribute__((ext_vector_type(4)))  int          i32x4;

// ------------- CDNA5 feature detection (compile-time) ----------------
#if defined(__has_builtin)
#  if __has_builtin(__builtin_amdgcn_global_load_async_to_lds_b32)
#    define HAVE_ASYNC_LDS 1
#  endif
#  if __has_builtin(__builtin_amdgcn_tensor_load_to_lds) && \
      __has_builtin(__builtin_amdgcn_s_wait_tensorcnt)
#    define HAVE_TDM 1
#  endif
#endif
#ifndef HAVE_ASYNC_LDS
#  define HAVE_ASYNC_LDS 0
#endif
#ifndef HAVE_TDM
#  define HAVE_TDM 0
#endif

// generic->AS1/AS3 typed-pointer casts via integer round trip: for LDS the
// generic address' low 32 bits are the LDS byte offset (ISA 10.2 aperture
// mapping). The async builtins take addrspace-qualified int* (clang prints
// them as "__device__ int *" / "__shared__ int *").
#define GPTR_I32(p) ((__attribute__((address_space(1))) int*)(unsigned long long)(p))
#define LPTR_I32(p) ((__attribute__((address_space(3))) int*)(unsigned int)(unsigned long long)(p))

// ---------------------------------------------------------------------
// Kernel 1: IoU + matcher. One block handles 256 proposals of one image.
// GT boxes (4KB tile) staged into LDS by the Tensor Data Mover (one D#
// issued by wave 0, completion via s_wait_tensorcnt); GT labels staged
// with async global->LDS loads (ASYNCcnt).
// ---------------------------------------------------------------------
__global__ void iou_match_kernel(const float* __restrict__ proposals,
                                 const float* __restrict__ gt_boxes,
                                 const int*   __restrict__ gt_labels,
                                 int* __restrict__ labels_all,
                                 int* __restrict__ matches_all) {
    const int b   = blockIdx.y;
    const int tid = threadIdx.x;

    __shared__ float4 sh_gtb[G];
    __shared__ int    sh_gtl[G];

    const float* gsrc = gt_boxes + (size_t)b * G * 4;

#if HAVE_TDM
    if (tid < 32) {  // one wave issues the DMA (EXEC ignored by TDM)
        unsigned long long ga = (unsigned long long)gsrc;
        unsigned int lds = (unsigned int)(unsigned long long)(&sh_gtb[0]);
        // D# group0: count=1 | lds_addr | global_addr[56:0] | type=2
        u32x4 g0;
        g0.x = 1u;                                   // count=1, user mode
        g0.y = lds;                                  // lds_addr (bytes)
        g0.z = (unsigned int)(ga & 0xFFFFFFFFu);     // global_addr[31:0]
        g0.w = (unsigned int)((ga >> 32) & 0x1FFFFFFu) | (2u << 30); // +type=2
        // D# group1: 2D tensor, data_size=4B, one row of 1024 f32
        i32x8 g1;
        g1.s0 = (int)(2u << 16);      // wg_mask=0, data_size=2 (4B)
        g1.s1 = (int)(1024u << 16);   // tensor_dim0[15:0]=1024 at bits 63:48
        g1.s2 = (int)(1u << 16);      // tensor_dim0 hi=0 | tensor_dim1=1
        g1.s3 = (int)(1024u << 16);   // tensor_dim1 hi=0 | tile_dim0=1024
        g1.s4 = 1;                    // tile_dim1=1, tile_dim2=0
        g1.s5 = 1024;                 // tensor_dim0_stride[31:0]
        g1.s6 = 0;                    // stride0 hi | stride1 lo
        g1.s7 = 0;
        i32x4 z4 = {0, 0, 0, 0};
        i32x8 z8 = {0, 0, 0, 0, 0, 0, 0, 0};
        __builtin_amdgcn_tensor_load_to_lds(g0, g1, z4, z4, z8, 0);
        __builtin_amdgcn_s_wait_tensorcnt(0);
    }
#else
    sh_gtb[tid] = ((const float4*)gsrc)[tid];
#endif

#if HAVE_ASYNC_LDS
    __builtin_amdgcn_global_load_async_to_lds_b32(GPTR_I32(gt_labels + b * G + tid),
                                                  LPTR_I32(&sh_gtl[tid]), 0, 0);
    asm volatile("s_wait_asynccnt 0x0" ::: "memory");
#else
    sh_gtl[tid] = gt_labels[b * G + tid];
#endif
    __syncthreads();

    const int j = blockIdx.x * 256 + tid;
    if (j >= M) return;

    float4 box;
    if (j < P) {
        const float* p = proposals + ((size_t)b * P + j) * 4;
        __builtin_prefetch(p + 256 * 4, 0, 1);   // global_prefetch_b8
        box = *(const float4*)p;
    } else {
        box = sh_gtb[j - P];
    }

    const float area_b = (box.z - box.x) * (box.w - box.y);
    float best = -1.0f;
    int   bi   = 0;
#pragma unroll 4
    for (int g = 0; g < G; ++g) {
        float4 a = sh_gtb[g];
        float area_a = (a.z - a.x) * (a.w - a.y);
        float lx = fmaxf(a.x, box.x);
        float ly = fmaxf(a.y, box.y);
        float rx = fminf(a.z, box.z);
        float ry = fminf(a.w, box.w);
        float w  = fmaxf(rx - lx, 0.0f);
        float h  = fmaxf(ry - ly, 0.0f);
        float inter = w * h;
        float iou = inter / (area_a + area_b - inter);
        if (iou > best) { best = iou; bi = g; }  // strict > == argmax-first
    }

    // FG_T == BG_T == 0.5 -> the "ignore" band is empty
    int lab = (best < BG_T) ? 0 : sh_gtl[bi];
    labels_all[b * M + j]  = lab;
    matches_all[b * M + j] = bi;
}

// ---------------------------------------------------------------------
// Kernel 2: balanced sampler. One block (8 wave32) per image; sequential
// chunk scan over M using wave32 ballots + cross-wave LDS prefix.
// Label chunks are double-buffered via async global->LDS loads: chunk
// k+1 is in flight while chunk k is scanned (s_wait_asynccnt 0x1).
// ---------------------------------------------------------------------
__global__ void sample_kernel(const int* __restrict__ labels_all,
                              const int* __restrict__ matches_all,
                              const float* __restrict__ proposals,
                              const float* __restrict__ gt_boxes,
                              int*   __restrict__ out_lbl,
                              int*   __restrict__ out_match,
                              float* __restrict__ out_box) {
    const int b    = blockIdx.x;
    const int tid  = threadIdx.x;
    const int lane = tid & 31;
    const int wave = tid >> 5;

    __shared__ int wp[8], wn[8], wsx[8];
    __shared__ int red[8];
    __shared__ int run_pos, run_neg, run_sel, sh_npos;
#if HAVE_ASYNC_LDS
    __shared__ int sh_lab[2][256];
#endif

    const int* lbl = labels_all + (size_t)b * M;
    const int* mat = matches_all + (size_t)b * M;

    // init outputs (defensive: ws is poisoned by harness)
    for (int s = tid; s < S; s += 256) {
        out_lbl[b * S + s]   = 0;
        out_match[b * S + s] = 0;
        float* o = out_box + ((size_t)b * S + s) * 4;
        o[0] = 0.f; o[1] = 0.f; o[2] = 1.f; o[3] = 1.f;
    }

    // total positive count -> n_pos = min(total, 128)
    int cp = 0;
    for (int j = tid; j < M; j += 256) cp += (lbl[j] > 0);
    for (int o = 16; o > 0; o >>= 1) cp += __shfl_xor(cp, o, 32);
    if (lane == 0) red[wave] = cp;
    __syncthreads();
    if (tid == 0) {
        int t = 0;
        for (int w = 0; w < 8; ++w) t += red[w];
        sh_npos = (t < NPOS_MAX) ? t : NPOS_MAX;
        run_pos = run_neg = run_sel = 0;
    }
    __syncthreads();
    const int n_pos = sh_npos;
    const int n_neg = S - n_pos;

#if HAVE_ASYNC_LDS
    // preload chunk 0
    if (tid < M)
        __builtin_amdgcn_global_load_async_to_lds_b32(GPTR_I32(lbl + tid),
                                                      LPTR_I32(&sh_lab[0][tid]), 0, 0);
#endif

    for (int base = 0; base < M; base += 256) {
        const int ch = base >> 8;
        const int j  = base + tid;

#if HAVE_ASYNC_LDS
        // issue next chunk into the other buffer
        const int jn = base + 256 + tid;
        const bool need_next = (jn < M);
        const bool wave_issued = (__ballot(need_next) != 0ull);  // wave-uniform
        if (need_next)
            __builtin_amdgcn_global_load_async_to_lds_b32(
                GPTR_I32(lbl + jn), LPTR_I32(&sh_lab[(ch + 1) & 1][tid]), 0, 0);
        // async loads complete in order: cnt<=1 => current chunk landed
        if (wave_issued) asm volatile("s_wait_asynccnt 0x1" ::: "memory");
        else             asm volatile("s_wait_asynccnt 0x0" ::: "memory");
        const int l = (j < M) ? sh_lab[ch & 1][tid] : -1;
#else
        const int l = (j < M) ? lbl[j] : -1;
#endif
        const bool p = (j < M) && (l > 0);
        const bool n = (j < M) && (l == 0);

        unsigned long long bp = __ballot(p);
        unsigned long long bn = __ballot(n);
        unsigned long long lm = (1ull << lane) - 1ull;
        int ppre = __popcll(bp & lm);
        int npre = __popcll(bn & lm);
        if (lane == 0) { wp[wave] = __popcll(bp); wn[wave] = __popcll(bn); }
        __syncthreads();
        int poff = 0, noff = 0;
        for (int w = 0; w < wave; ++w) { poff += wp[w]; noff += wn[w]; }

        const int prank = run_pos + poff + ppre;  // 0-based rank among positives
        const int nrank = run_neg + noff + npre;
        const bool sel = (p && prank < n_pos) || (n && nrank < n_neg);

        unsigned long long bs = __ballot(sel);
        int spre = __popcll(bs & lm);
        if (lane == 0) wsx[wave] = __popcll(bs);
        __syncthreads();
        int soff = 0;
        for (int w = 0; w < wave; ++w) soff += wsx[w];

        if (sel) {
            const int out = run_sel + soff + spre;  // index-sorted position
            if (out < S) {
                out_lbl[b * S + out]   = l;
                out_match[b * S + out] = mat[j];
                const float* src = (j < P)
                    ? (proposals + ((size_t)b * P + j) * 4)
                    : (gt_boxes  + ((size_t)b * G + (j - P)) * 4);
                float* o = out_box + ((size_t)b * S + out) * 4;
                o[0] = src[0]; o[1] = src[1]; o[2] = src[2]; o[3] = src[3];
            }
        }
        __syncthreads();
        if (tid == 0) {
            int tp = 0, tn = 0, ts = 0;
            for (int w = 0; w < 8; ++w) { tp += wp[w]; tn += wn[w]; ts += wsx[w]; }
            run_pos += tp; run_neg += tn; run_sel += ts;
        }
        __syncthreads();
    }
}

// ---------------------------------------------------------------------
// Kernel 3: per-row CE + box encode + smooth-L1 contributions.
// ---------------------------------------------------------------------
__global__ void loss_kernel(const float* __restrict__ logits,
                            const float* __restrict__ boxreg,
                            const float* __restrict__ gt_boxes,
                            const float* __restrict__ gt_thetas,
                            const int*   __restrict__ s_lbl,
                            const int*   __restrict__ s_match,
                            const float* __restrict__ s_box,
                            float* __restrict__ cls_c,
                            float* __restrict__ box_c) {
    const int i = blockIdx.x * blockDim.x + threadIdx.x;
    if (i >= B * S) return;
    const int b = i / S;

    int l = s_lbl[i];
    if (l < 0) l = 0;
    if (l >= C) l = C - 1;

    // log-softmax CE
    const float* row = logits + (size_t)i * C;
    float mx = row[0];
    for (int c = 1; c < C; ++c) mx = fmaxf(mx, row[c]);
    float se = 0.0f;
    for (int c = 0; c < C; ++c) se += __expf(row[c] - mx);
    cls_c[i] = -(row[l] - mx - __logf(se));

    // box encode (weights 10,10,5,5) + theta target
    int m = s_match[i];
    if (m < 0) m = 0;
    if (m >= G) m = G - 1;
    const float* g  = gt_boxes + ((size_t)b * G + m) * 4;
    const float* pr = s_box + (size_t)i * 4;
    float pw = pr[2] - pr[0], ph = pr[3] - pr[1];
    float px = pr[0] + 0.5f * pw, py = pr[1] + 0.5f * ph;
    float gw = g[2] - g[0], gh = g[3] - g[1];
    float gx = g[0] + 0.5f * gw, gy = g[1] + 0.5f * gh;
    float t[5];
    t[0] = 10.0f * (gx - px) / pw;
    t[1] = 10.0f * (gy - py) / ph;
    t[2] = 5.0f * __logf(gw / pw);
    t[3] = 5.0f * __logf(gh / ph);
    t[4] = gt_thetas[b * G + m];

    const float* pd = boxreg + (size_t)i * (C * 5) + l * 5;
    float sl = 0.0f;
#pragma unroll
    for (int k = 0; k < 5; ++k) {
        float d  = pd[k] - t[k];
        float ad = fabsf(d);
        sl += (ad < BETA) ? 0.5f * d * d / BETA : ad - 0.5f * BETA;
    }
    box_c[i] = (l > 0) ? sl : 0.0f;
}

// ---------------------------------------------------------------------
// Kernel 4: final reduction. 8192 -> 256 -> 64 partials, then finish on
// the matrix pipe: v_wmma_f32_16x16x32_f16 with A = ones. With A all-ones
// D[m,n] = sum_k B[k,n], so sum(D) = 16 * sum(B-slots); the 64 partials
// occupy elements {0,1} of each lane's B fragment, rest zero.
// ---------------------------------------------------------------------
__global__ void wmma_reduce_kernel(const float* __restrict__ cls_c,
                                   const float* __restrict__ box_c,
                                   float* __restrict__ out) {
    __shared__ float pc[256], pb[256];
    __shared__ float rc[64], rb[64];
    __shared__ float fin0, fin1;
    const int tid = threadIdx.x;

    float sc = 0.0f, sb = 0.0f;
    for (int i = tid; i < B * S; i += 256) { sc += cls_c[i]; sb += box_c[i]; }
    pc[tid] = sc; pb[tid] = sb;
    __syncthreads();
    if (tid < 64) {
        rc[tid] = pc[tid] + pc[tid + 64] + pc[tid + 128] + pc[tid + 192];
        rb[tid] = pb[tid] + pb[tid + 64] + pb[tid + 128] + pb[tid + 192];
    }
    __syncthreads();

    if (tid < 32) {  // wave 0 only: EXEC all-ones as WMMA requires
        v16h a, bc, bb;
#pragma unroll
        for (int e = 0; e < 16; ++e) a[e] = (_Float16)1.0f;
#pragma unroll
        for (int e = 0; e < 16; ++e) {
            float vc = (e < 2) ? rc[tid * 2 + e] : 0.0f;
            float vb = (e < 2) ? rb[tid * 2 + e] : 0.0f;
            bc[e] = (_Float16)vc;
            bb[e] = (_Float16)vb;
        }
        v8f z = {};
        v8f dc = __builtin_amdgcn_wmma_f32_16x16x32_f16(false, a, false, bc,
                                                        (short)0, z, false, false);
        v8f db = __builtin_amdgcn_wmma_f32_16x16x32_f16(false, a, false, bb,
                                                        (short)0, z, false, false);
        float tc = 0.0f, tb = 0.0f;
#pragma unroll
        for (int k = 0; k < 8; ++k) { tc += dc[k]; tb += db[k]; }
        for (int o = 16; o > 0; o >>= 1) {
            tc += __shfl_xor(tc, o, 32);
            tb += __shfl_xor(tb, o, 32);
        }
        if (tid == 0) { fin0 = tc * (1.0f / 16.0f); fin1 = tb * (1.0f / 16.0f); }
    }
    __syncthreads();
    if (tid == 0) {
        out[0] = fin0 / (float)(B * S);  // mean CE
        out[1] = fin1 / (float)(B * S);  // box loss / N
    }
}

// ---------------------------------------------------------------------
extern "C" void kernel_launch(void* const* d_in, const int* in_sizes, int n_in,
                              void* d_out, int out_size, void* d_ws, size_t ws_size,
                              hipStream_t stream) {
    const float* class_logits   = (const float*)d_in[0];  // [B*S, 91]
    const float* box_regression = (const float*)d_in[1];  // [B*S, 455]
    const float* proposals      = (const float*)d_in[2];  // [B, P, 4]
    const float* gt_boxes       = (const float*)d_in[3];  // [B, G, 4]
    const float* gt_thetas      = (const float*)d_in[4];  // [B, G]
    const int*   gt_labels      = (const int*)d_in[5];    // [B, G]

    char* ws = (char*)d_ws;
    int*   labels_all  = (int*)ws;   ws += sizeof(int) * B * M;
    int*   matches_all = (int*)ws;   ws += sizeof(int) * B * M;
    int*   s_lbl       = (int*)ws;   ws += sizeof(int) * B * S;
    int*   s_match     = (int*)ws;   ws += sizeof(int) * B * S;
    float* s_box       = (float*)ws; ws += sizeof(float) * B * S * 4;
    float* cls_c       = (float*)ws; ws += sizeof(float) * B * S;
    float* box_c       = (float*)ws; ws += sizeof(float) * B * S;

    iou_match_kernel<<<dim3((M + 255) / 256, B), 256, 0, stream>>>(
        proposals, gt_boxes, gt_labels, labels_all, matches_all);

    sample_kernel<<<B, 256, 0, stream>>>(
        labels_all, matches_all, proposals, gt_boxes, s_lbl, s_match, s_box);

    loss_kernel<<<(B * S + 255) / 256, 256, 0, stream>>>(
        class_logits, box_regression, gt_boxes, gt_thetas,
        s_lbl, s_match, s_box, cls_c, box_c);

    wmma_reduce_kernel<<<1, 256, 0, stream>>>(cls_c, box_c, (float*)d_out);
}
